// GraphNetwork_14336600834642
// MI455X (gfx1250) — matrix operand order repaired and checked by reading, not compiled, not run on hardware
//
#include <hip/hip_runtime.h>
#include <hip/hip_bf16.h>

typedef _Float16 v8h  __attribute__((ext_vector_type(8)));
typedef _Float16 v16h __attribute__((ext_vector_type(16)));
typedef float    v8f  __attribute__((ext_vector_type(8)));

#define N_ATOMS 200000
#define N_BONDS 400000
#define N_MOLS  8192
#define NLAYERS 4

// ---------- helpers ----------
__device__ __forceinline__ v16h ld_frag2(const _Float16* p0, const _Float16* p1) {
    v8h lo = *(const v8h*)p0;
    v8h hi = *(const v8h*)p1;
    return __builtin_shufflevector(lo, hi, 0,1,2,3,4,5,6,7,8,9,10,11,12,13,14,15);
}
// fast sigmoid: v_exp_f32 + v_add_f32 + v_rcp_f32 (no precise-div chain)
__device__ __forceinline__ float sigmoidf_fast(float x) {
    return __builtin_amdgcn_rcpf(1.0f + __expf(-x));
}

// ---------- embedding gather: out[i*64+d] = emb[types[i]*64+d] ----------
__global__ void embed64_kernel(const float* __restrict__ emb, const int* __restrict__ types,
                               float* __restrict__ out, int n) {
    int idx = blockIdx.x * blockDim.x + threadIdx.x;
    if (idx >= n * 64) return;
    int i = idx >> 6, d = idx & 63;
    out[idx] = emb[types[i] * 64 + d];
}

// ---------- batchnorm (inference) + cast to f16 ----------
__global__ void bn_f16_kernel(const float* __restrict__ state,
                              const float* __restrict__ gamma, const float* __restrict__ beta,
                              const float* __restrict__ mean,  const float* __restrict__ var,
                              _Float16* __restrict__ out, int n) {
    int idx = blockIdx.x * blockDim.x + threadIdx.x;
    if (idx >= n * 64) return;
    int d = idx & 63;
    float x = state[idx];
    float y = gamma[d] * (x - mean[d]) * rsqrtf(var[d] + 1e-3f) + beta[d];
    out[idx] = (_Float16)y;
}

// ---------- pack KxN row-major f32 weight into WMMA B-fragment layout f16 ----------
// dst[((kt*(N/16)+nt)*32 + lane)*16 + e] = W[kt*32 + (e>>3)*16 + (lane>>4)*8 + (e&7)][nt*16 + (lane&15)]
__global__ void pack_w_kernel(const float* __restrict__ src, _Float16* __restrict__ dst,
                              int K, int N) {
    int idx = blockIdx.x * blockDim.x + threadIdx.x;
    if (idx >= K * N) return;
    int e    = idx & 15;
    int lane = (idx >> 4) & 31;
    int frag = idx >> 9;
    int ntiles = N >> 4;
    int nt = frag % ntiles;
    int kt = frag / ntiles;
    int k = kt * 32 + (e >> 3) * 16 + (lane >> 4) * 8 + (e & 7);
    int n = nt * 16 + (lane & 15);
    dst[idx] = (_Float16)src[k * N + n];
}

// ---------- main per-layer bond update: WMMA matmuls + atomic message scatter ----------
// LDS (halfs): W1p[24576] W2p[8192] Wap[4096] | per-wave A[16x192] H1[16x128]
__global__ void __launch_bounds__(128)
bond_update_kernel(const _Float16* __restrict__ w1p, const _Float16* __restrict__ w2p,
                   const _Float16* __restrict__ wap, const float* __restrict__ bias2,
                   const _Float16* __restrict__ a_bn, const _Float16* __restrict__ b_bn,
                   const int* __restrict__ conn,
                   float* __restrict__ bond_state, float* __restrict__ atom_state,
                   int num_tiles) {
    extern __shared__ _Float16 smem[];
    _Float16* sW1 = smem;            // 24576 halfs
    _Float16* sW2 = smem + 24576;    // 8192
    _Float16* sWa = smem + 32768;    // 4096
    const int tid  = threadIdx.x;
    const int wave = tid >> 5;
    const int lane = tid & 31;
    _Float16* sA  = smem + 36864 + wave * 3072;   // 16 x 192
    _Float16* sH1 = smem + 49152 + wave * 2048;   // 16 x 128

    // cooperative weight load (global -> LDS), once per block
    {
        const uint4* s1 = (const uint4*)w1p; uint4* d1 = (uint4*)sW1;
        for (int i = tid; i < 3072; i += 128) d1[i] = s1[i];
        const uint4* s2 = (const uint4*)w2p; uint4* d2 = (uint4*)sW2;
        for (int i = tid; i < 1024; i += 128) d2[i] = s2[i];
        const uint4* s3 = (const uint4*)wap; uint4* d3 = (uint4*)sWa;
        for (int i = tid; i < 512;  i += 128) d3[i] = s3[i];
    }
    __syncthreads();

    const int m16 = lane & 15;
    const int hi  = lane >> 4;
    const int totalWaves = gridDim.x * 4;

    for (int tile = blockIdx.x * 4 + wave; tile < num_tiles; tile += totalWaves) {
        const int bond0 = tile * 16;

        // ---- gather A panel: [src(64) | tgt(64) | bond(64)] f16 rows ----
        {
            int bond = bond0 + m16;
            int tgti = conn[bond * 2 + 0];
            int srci = conn[bond * 2 + 1];
            if (hi == 0) {
                const uint4* s = (const uint4*)(a_bn + (size_t)srci * 64);
                uint4* d = (uint4*)(sA + m16 * 192);
                #pragma unroll
                for (int j = 0; j < 8; ++j) d[j] = s[j];
                const uint4* sb = (const uint4*)(b_bn + (size_t)bond * 64);
                uint4* db = (uint4*)(sA + m16 * 192 + 128);
                #pragma unroll
                for (int j = 0; j < 8; ++j) db[j] = sb[j];
            } else {
                const uint4* t = (const uint4*)(a_bn + (size_t)tgti * 64);
                uint4* d = (uint4*)(sA + m16 * 192 + 64);
                #pragma unroll
                for (int j = 0; j < 8; ++j) d[j] = t[j];
            }
        }
        asm volatile("s_wait_dscnt 0" ::: "memory");   // wave-local LDS store->load ordering

        // ---- matmul1: (16x192) @ (192x128), sigmoid -> sH1 ----
        #pragma unroll
        for (int nt = 0; nt < 8; ++nt) {
            v8f acc = {};
            #pragma unroll
            for (int kc = 0; kc < 6; ++kc) {
                v16h a = ld_frag2(sA + m16 * 192 + kc * 32 + hi * 8,
                                  sA + m16 * 192 + kc * 32 + 16 + hi * 8);
                const _Float16* bp = sW1 + ((kc * 8 + nt) * 32 + lane) * 16;
                v16h b = ld_frag2(bp, bp + 8);
                acc = __builtin_amdgcn_wmma_f32_16x16x32_f16(false, a, false, b,
                                                             (short)0, acc, false, false);
            }
            #pragma unroll
            for (int i = 0; i < 8; ++i)
                sH1[(i + 8 * hi) * 128 + nt * 16 + m16] = (_Float16)sigmoidf_fast(acc[i]);
        }
        asm volatile("s_wait_dscnt 0" ::: "memory");

        // ---- matmul2: (16x128) @ (128x64) + bias -> h2 regs ----
        float h2r[4][8];
        #pragma unroll
        for (int nt = 0; nt < 4; ++nt) {
            v8f acc = {};
            #pragma unroll
            for (int kc = 0; kc < 4; ++kc) {
                v16h a = ld_frag2(sH1 + m16 * 128 + kc * 32 + hi * 8,
                                  sH1 + m16 * 128 + kc * 32 + 16 + hi * 8);
                const _Float16* bp = sW2 + ((kc * 4 + nt) * 32 + lane) * 16;
                v16h b = ld_frag2(bp, bp + 8);
                acc = __builtin_amdgcn_wmma_f32_16x16x32_f16(false, a, false, b,
                                                             (short)0, acc, false, false);
            }
            float bias = bias2[nt * 16 + m16];
            #pragma unroll
            for (int i = 0; i < 8; ++i) h2r[nt][i] = acc[i] + bias;
        }

        // ---- sa = sigmoid(src_atom @ W_au): A = first 64 cols of sA ----
        float sar[4][8];
        #pragma unroll
        for (int nt = 0; nt < 4; ++nt) {
            v8f acc = {};
            #pragma unroll
            for (int kc = 0; kc < 2; ++kc) {
                v16h a = ld_frag2(sA + m16 * 192 + kc * 32 + hi * 8,
                                  sA + m16 * 192 + kc * 32 + 16 + hi * 8);
                const _Float16* bp = sWa + ((kc * 4 + nt) * 32 + lane) * 16;
                v16h b = ld_frag2(bp, bp + 8);
                acc = __builtin_amdgcn_wmma_f32_16x16x32_f16(false, a, false, b,
                                                             (short)0, acc, false, false);
            }
            #pragma unroll
            for (int i = 0; i < 8; ++i) sar[nt][i] = sigmoidf_fast(acc[i]);
        }

        // ---- writeback: bond_state += nb ; atomic scatter sa*nb into atom_state ----
        int tgt_row[8];
        #pragma unroll
        for (int i = 0; i < 8; ++i) tgt_row[i] = conn[(bond0 + i + 8 * hi) * 2];
        #pragma unroll
        for (int i = 0; i < 8; ++i) {
            int bond = bond0 + i + 8 * hi;
            #pragma unroll
            for (int nt = 0; nt < 4; ++nt) {
                int n = nt * 16 + m16;
                float nbv = h2r[nt][i];
                size_t bofs = (size_t)bond * 64 + n;
                bond_state[bofs] = bond_state[bofs] + nbv;
                atomicAdd(atom_state + (size_t)tgt_row[i] * 64 + n, sar[nt][i] * nbv);
            }
        }
        asm volatile("" ::: "memory"); // keep panel reads before next-iter overwrites
    }
}

// ---------- molecule segment sum ----------
__global__ void zero_f32_kernel(float* __restrict__ p, int n) {
    int i = blockIdx.x * blockDim.x + threadIdx.x;
    if (i < n) p[i] = 0.0f;
}
__global__ void mol_accum_kernel(const float* __restrict__ atom_state,
                                 const int* __restrict__ nga, float* __restrict__ mol) {
    int idx = blockIdx.x * blockDim.x + threadIdx.x;
    if (idx >= N_ATOMS * 64) return;
    int a = idx >> 6, d = idx & 63;
    atomicAdd(&mol[(size_t)nga[a] * 64 + d], atom_state[idx]);
}

// ---------- readout MLP: 64 -> relu 128 -> relu 64 -> 1 ----------
__global__ void __launch_bounds__(128)
readout_kernel(const float* __restrict__ mol,
               const float* __restrict__ Wo1, const float* __restrict__ bo1,
               const float* __restrict__ Wo2, const float* __restrict__ bo2,
               const float* __restrict__ Wl,  const float* __restrict__ bl,
               float* __restrict__ out) {
    __shared__ float sm[64];
    __shared__ float sh1[128];
    __shared__ float sh2[64];
    int m = blockIdx.x;
    int t = threadIdx.x;
    if (t < 64) sm[t] = mol[(size_t)m * 64 + t];
    __syncthreads();
    float a1 = bo1[t];
    for (int k = 0; k < 64; ++k) a1 += sm[k] * Wo1[k * 128 + t];
    sh1[t] = fmaxf(a1, 0.0f);
    __syncthreads();
    if (t < 64) {
        float a2 = bo2[t];
        for (int k = 0; k < 128; ++k) a2 += sh1[k] * Wo2[k * 64 + t];
        sh2[t] = fmaxf(a2, 0.0f);
    }
    __syncthreads();
    if (t == 0) {
        float a3 = bl[0];
        for (int k = 0; k < 64; ++k) a3 += sh2[k] * Wl[k];
        out[m] = a3;
    }
}

extern "C" void kernel_launch(void* const* d_in, const int* in_sizes, int n_in,
                              void* d_out, int out_size, void* d_ws, size_t ws_size,
                              hipStream_t stream) {
    const float* atom_emb   = (const float*)d_in[0];
    const float* bond_emb   = (const float*)d_in[1];
    const float* bn_a_gamma = (const float*)d_in[2];
    const float* bn_a_beta  = (const float*)d_in[3];
    const float* bn_a_mean  = (const float*)d_in[4];
    const float* bn_a_var   = (const float*)d_in[5];
    const float* bn_b_gamma = (const float*)d_in[6];
    const float* bn_b_beta  = (const float*)d_in[7];
    const float* bn_b_mean  = (const float*)d_in[8];
    const float* bn_b_var   = (const float*)d_in[9];
    const float* W_bu1      = (const float*)d_in[10];
    const float* W_bu2      = (const float*)d_in[11];
    const float* b_bu2      = (const float*)d_in[12];
    const float* W_au       = (const float*)d_in[13];
    const float* W_o1       = (const float*)d_in[14];
    const float* b_o1       = (const float*)d_in[15];
    const float* W_o2       = (const float*)d_in[16];
    const float* b_o2       = (const float*)d_in[17];
    const float* W_last     = (const float*)d_in[18];
    const float* b_last     = (const float*)d_in[19];
    const int* atom_types   = (const int*)d_in[20];
    const int* bond_types   = (const int*)d_in[21];
    const int* nga          = (const int*)d_in[22];
    const int* conn         = (const int*)d_in[23];
    float* out = (float*)d_out;

    // workspace carve-up
    char* ws = (char*)d_ws;
    float*    atom_state = (float*)ws;                               // N_ATOMS*64 f32
    float*    bond_state = (float*)(ws + (size_t)N_ATOMS * 64 * 4);  // N_BONDS*64 f32
    _Float16* a_bn = (_Float16*)((char*)bond_state + (size_t)N_BONDS * 64 * 4);
    _Float16* b_bn = a_bn + (size_t)N_ATOMS * 64;
    _Float16* w1p  = b_bn + (size_t)N_BONDS * 64;       // NLAYERS * 24576
    _Float16* w2p  = w1p + NLAYERS * 24576;             // NLAYERS * 8192
    _Float16* wap  = w2p + NLAYERS * 8192;              // NLAYERS * 4096
    float*    mol  = (float*)(wap + NLAYERS * 4096);    // N_MOLS*64 f32

    const int T = 256;

    // embeddings -> initial states
    embed64_kernel<<<(N_ATOMS * 64 + T - 1) / T, T, 0, stream>>>(atom_emb, atom_types, atom_state, N_ATOMS);
    embed64_kernel<<<(N_BONDS * 64 + T - 1) / T, T, 0, stream>>>(bond_emb, bond_types, bond_state, N_BONDS);

    // pack weights into WMMA B-fragment layout (f16)
    for (int l = 0; l < NLAYERS; ++l) {
        pack_w_kernel<<<(192 * 128 + T - 1) / T, T, 0, stream>>>(W_bu1 + (size_t)l * 192 * 128, w1p + l * 24576, 192, 128);
        pack_w_kernel<<<(128 * 64 + T - 1) / T, T, 0, stream>>>(W_bu2 + (size_t)l * 128 * 64, w2p + l * 8192, 128, 64);
        pack_w_kernel<<<(64 * 64 + T - 1) / T, T, 0, stream>>>(W_au + (size_t)l * 64 * 64, wap + l * 4096, 64, 64);
    }

    // 112 KB dynamic LDS per block (CDNA5 WGP has 320 KB)
    const size_t ldsBytes = 57344 * sizeof(_Float16);
    hipFuncSetAttribute((const void*)bond_update_kernel,
                        hipFuncAttributeMaxDynamicSharedMemorySize, (int)ldsBytes);

    const int numTiles = N_BONDS / 16; // 25000
    for (int l = 0; l < NLAYERS; ++l) {
        bn_f16_kernel<<<(N_ATOMS * 64 + T - 1) / T, T, 0, stream>>>(
            atom_state, bn_a_gamma + l * 64, bn_a_beta + l * 64,
            bn_a_mean + l * 64, bn_a_var + l * 64, a_bn, N_ATOMS);
        bn_f16_kernel<<<(N_BONDS * 64 + T - 1) / T, T, 0, stream>>>(
            bond_state, bn_b_gamma + l * 64, bn_b_beta + l * 64,
            bn_b_mean + l * 64, bn_b_var + l * 64, b_bn, N_BONDS);
        bond_update_kernel<<<1024, 128, ldsBytes, stream>>>(
            w1p + l * 24576, w2p + l * 8192, wap + l * 4096, b_bu2 + l * 64,
            a_bn, b_bn, conn, bond_state, atom_state, numTiles);
    }

    // molecule readout
    zero_f32_kernel<<<(N_MOLS * 64 + T - 1) / T, T, 0, stream>>>(mol, N_MOLS * 64);
    mol_accum_kernel<<<(N_ATOMS * 64 + T - 1) / T, T, 0, stream>>>(atom_state, nga, mol);
    readout_kernel<<<N_MOLS, 128, 0, stream>>>(mol, W_o1, b_o1, W_o2, b_o2, W_last, b_last, out);
}
